// MultiHeadAttention_35871566856479
// MI455X (gfx1250) — matrix operand order repaired and checked by reading, not compiled, not run on hardware
//
#include <hip/hip_runtime.h>
#include <hip/hip_bf16.h>

// ---------------------------------------------------------------------------
// MHA forward for MI455X (gfx1250, wave32):
//   q = X_q Wq^T ; k = X_k Wk^T ; v = X_v Wv^T   (f16 WMMA, f32 accum)
//   flash attention (online softmax, additive mask, async double-buffered
//   K/V staging via GLOBAL_LOAD_ASYNC_TO_LDS_B128)  -> ctx (f16)
//   out = ctx Wo^T                                  (f32)
// Workspace layout (bytes):
//   [0,8M)    Wq16,Wk16,Wv16,Wo16 (f16, row-major (n,k) == B^T for x@W^T)
//   [8M,24M)  Q16   (B,H,S,DH) f16
//   [24M,40M) K16   (B,H,S,DH) f16
//   [40M,56M) VT16  (B,H,DH,S) f16   (transposed so PV B-frags are contiguous)
//   [56M,72M) CTX16 (B,S,D)    f16
//   [72M,..)  maskAdd (B*S) f32 : -1e9 where masked else 0
// ---------------------------------------------------------------------------

typedef _Float16 half_t;
typedef __attribute__((ext_vector_type(16))) _Float16 v16h;
typedef __attribute__((ext_vector_type(8)))  _Float16 v8h;
typedef __attribute__((ext_vector_type(8)))  float    v8f;
typedef int v4i __attribute__((vector_size(4 * sizeof(int))));

#define D_  1024
#define S_  2048
#define H_  16
#define DH_ 64
#define B_  4
#define BS_ (B_ * S_)   // 8192 rows

union FragU { v16h v; v8h h[2]; };

__device__ __forceinline__ v16h frag2(const half_t* p0, const half_t* p1) {
  FragU u;
  u.h[0] = *(const v8h*)p0;
  u.h[1] = *(const v8h*)p1;
  return u.v;
}

__device__ __forceinline__ v8f wmma_f16(v16h a, v16h b, v8f c) {
  // D = A(16x32 f16) * B(32x16 f16) + C(16x16 f32)
  return __builtin_amdgcn_wmma_f32_16x16x32_f16(false, a, false, b, (short)0, c,
                                                false, false);
}

__device__ __forceinline__ v8f zero8() {
  v8f z;
#pragma unroll
  for (int e = 0; e < 8; ++e) z[e] = 0.0f;
  return z;
}

__device__ __forceinline__ float rmax16(float v) {
#pragma unroll
  for (int m = 1; m < 16; m <<= 1) v = fmaxf(v, __shfl_xor(v, m, 32));
  return v;
}

__device__ __forceinline__ float rsum16(float v) {
#pragma unroll
  for (int m = 1; m < 16; m <<= 1) v += __shfl_xor(v, m, 32);
  return v;
}

// ---------------------------------------------------------------------------
// CDNA5 async global->LDS copy (ASYNCcnt-tracked, no VGPR round-trip).
// Builtin signature (from hipcc diagnostic): param1 = v4i addrspace(1)* (global
// src), param2 = v4i addrspace(3)* (LDS dst), then imm offset, imm cpol.
// Falls back to a synchronous copy if the builtin is unavailable.
// ---------------------------------------------------------------------------
#if __has_builtin(__builtin_amdgcn_global_load_async_to_lds_b128)
#define HAS_ASYNC_LDS 1
#else
#define HAS_ASYNC_LDS 0
#endif

typedef __attribute__((address_space(1))) v4i* as1_v4i_p;
typedef __attribute__((address_space(3))) v4i* as3_v4i_p;

__device__ __forceinline__ void async_copy16(const half_t* g, half_t* l) {
#if HAS_ASYNC_LDS
  __builtin_amdgcn_global_load_async_to_lds_b128((as1_v4i_p)g, (as3_v4i_p)l,
                                                 0, 0);
#else
  *(v8h*)l = *(const v8h*)g;
#endif
}

__device__ __forceinline__ void async_wait0() {
#if HAS_ASYNC_LDS
  asm volatile("s_wait_asynccnt 0x0" ::: "memory");
#endif
}

// ---------------------------------------------------------------------------
// Elementwise converters
// ---------------------------------------------------------------------------
__global__ __launch_bounds__(256) void cvt_f32_to_f16(const float* __restrict__ src,
                                                      half_t* __restrict__ dst,
                                                      int n4) {
  int i = blockIdx.x * blockDim.x + threadIdx.x;
  if (i < n4) {
    float4 f = ((const float4*)src)[i];
    half_t* d = dst + (size_t)i * 4;
    d[0] = (half_t)f.x; d[1] = (half_t)f.y; d[2] = (half_t)f.z; d[3] = (half_t)f.w;
  }
}

__global__ __launch_bounds__(256) void cvt_mask(const unsigned char* __restrict__ m,
                                                float* __restrict__ ma, int n) {
  int i = blockIdx.x * blockDim.x + threadIdx.x;
  if (i < n) ma[i] = m[i] ? -1.0e9f : 0.0f;
}

// ---------------------------------------------------------------------------
// Projection GEMM: Out[m][n] = sum_k X[m][k] * W[n][k]
//   X fp32 (BS_ x D_), W f16 (D_ x D_ row-major == B^T), out f16.
//   mode 0/1: out[((b*H+h)*S+s)*DH+dh]   (Q / K layout)
//   mode 2 : out[((b*H+h)*DH+dh)*S+s]    (V transposed)
// 128x128 C tile, 8 waves, wave tile 32x64, BK=32.
// ---------------------------------------------------------------------------
__global__ __launch_bounds__(256) void proj_kernel(const float* __restrict__ X,
                                                   const half_t* __restrict__ W,
                                                   half_t* __restrict__ out,
                                                   int mode) {
  __shared__ half_t As[128][40];  // m-local x k-local (80B stride, mult of 16B)
  __shared__ half_t Bs[128][40];  // n-local x k-local (B^T tile)

  const int tid = threadIdx.x;
  const int lane = tid & 31, wave = tid >> 5;
  const int c = lane & 15, hl = lane >> 4;
  const int wm = wave & 3, wn = wave >> 2;           // 4x2 wave grid
  const int mb = blockIdx.y * 128, nb = blockIdx.x * 128;

  v8f acc[2][4];
#pragma unroll
  for (int i = 0; i < 2; ++i)
#pragma unroll
    for (int t = 0; t < 4; ++t) acc[i][t] = zero8();

  for (int k0 = 0; k0 < D_; k0 += 32) {
    __syncthreads();
    // A: 128x32 fp32 -> f16 into LDS (1024 float4 chunks, 4/thread)
#pragma unroll
    for (int i = 0; i < 4; ++i) {
      int idx = tid + i * 256;
      int r = idx >> 3, ch = idx & 7;
      float4 f = *(const float4*)(X + (size_t)(mb + r) * D_ + k0 + ch * 4);
      As[r][ch * 4 + 0] = (half_t)f.x;
      As[r][ch * 4 + 1] = (half_t)f.y;
      As[r][ch * 4 + 2] = (half_t)f.z;
      As[r][ch * 4 + 3] = (half_t)f.w;
    }
    // B: 128x32 f16 via async copy (512 8-elt chunks, 2/thread)
#pragma unroll
    for (int i = 0; i < 2; ++i) {
      int idx = tid + i * 256;
      int r = idx >> 2, ch = idx & 3;
      async_copy16(W + (size_t)(nb + r) * D_ + k0 + ch * 8, &Bs[r][ch * 8]);
    }
    async_wait0();
    __syncthreads();

    v16h af[2], bf[4];
#pragma unroll
    for (int i = 0; i < 2; ++i) {
      int r = wm * 32 + i * 16 + c;  // A-frag: row = lane%16
      af[i] = frag2(&As[r][8 * hl], &As[r][16 + 8 * hl]);
    }
#pragma unroll
    for (int t = 0; t < 4; ++t) {
      int r = wn * 64 + t * 16 + c;  // B-frag: col = lane%16, K=8*hl..+7, 16+8*hl..+7
      bf[t] = frag2(&Bs[r][8 * hl], &Bs[r][16 + 8 * hl]);
    }
#pragma unroll
    for (int i = 0; i < 2; ++i)
#pragma unroll
      for (int t = 0; t < 4; ++t) acc[i][t] = wmma_f16(af[i], bf[t], acc[i][t]);
  }

  // Epilogue: C/D layout row = 8*hl + j, col = lane%16.  Mode hoisted.
  if (mode != 2) {
#pragma unroll
    for (int i = 0; i < 2; ++i)
#pragma unroll
      for (int t = 0; t < 4; ++t)
#pragma unroll
        for (int j = 0; j < 8; ++j) {
          int rg = mb + wm * 32 + i * 16 + 8 * hl + j;  // m -> (b,s)
          int cg = nb + wn * 64 + t * 16 + c;           // n -> (h,dh)
          int b = rg >> 11, s = rg & (S_ - 1);
          int h = cg >> 6, dh = cg & 63;
          out[((size_t)((b * H_ + h) * S_ + s)) * DH_ + dh] = (half_t)acc[i][t][j];
        }
  } else {
#pragma unroll
    for (int i = 0; i < 2; ++i)
#pragma unroll
      for (int t = 0; t < 4; ++t)
#pragma unroll
        for (int j = 0; j < 8; ++j) {
          int rg = mb + wm * 32 + i * 16 + 8 * hl + j;
          int cg = nb + wn * 64 + t * 16 + c;
          int b = rg >> 11, s = rg & (S_ - 1);
          int h = cg >> 6, dh = cg & 63;
          out[((size_t)((b * H_ + h) * DH_ + dh)) * S_ + s] = (half_t)acc[i][t][j];
        }
  }
}

// ---------------------------------------------------------------------------
// Flash attention: per (b,h), 64 queries/block, 4 waves x 16 rows,
// 64-key blocks, online softmax, additive mask.
// K/V tiles double-buffered in LDS, filled by async global->LDS copies:
// while block i is computed, block i+1 streams in (one barrier per block;
// each wave drains its own ASYNCcnt before the barrier).
// ---------------------------------------------------------------------------
__global__ __launch_bounds__(128) void attn_kernel(const half_t* __restrict__ Q,
                                                   const half_t* __restrict__ K,
                                                   const half_t* __restrict__ VT,
                                                   const float* __restrict__ MA,
                                                   half_t* __restrict__ CTX) {
  __shared__ half_t Qs[64][72];         // queries x dh
  __shared__ half_t Ks[2][64][72];      // keys x dh   (== B^T for QK^T)
  __shared__ half_t Vs[2][64][72];      // dh x keys   (== B^T for PV)
  __shared__ half_t Ps[4][16][72];      // per-wave probs, rows x local-key

  const int tid = threadIdx.x;
  const int lane = tid & 31, wave = tid >> 5;
  const int c = lane & 15, hl = lane >> 4;
  const int bh = blockIdx.y;
  const int b = bh >> 4, h = bh & 15;
  const int s0 = blockIdx.x * 64;

  const half_t* kbase = K + (size_t)bh * S_ * DH_;
  const half_t* vbase = VT + (size_t)bh * DH_ * S_;

  // Kick off async K/V loads for block 0 into buffer 0.
#pragma unroll
  for (int i = 0; i < 4; ++i) {
    int idx = tid + i * 128;
    int r = idx >> 3, ch = idx & 7;
    async_copy16(kbase + (size_t)r * DH_ + ch * 8, &Ks[0][r][ch * 8]);
    async_copy16(vbase + (size_t)r * S_ + ch * 8, &Vs[0][r][ch * 8]);
  }

  // Load Q block (64x64 f16, 512 chunks, 4/thread) synchronously.
  const half_t* qb = Q + ((size_t)bh * S_ + s0) * DH_;
#pragma unroll
  for (int i = 0; i < 4; ++i) {
    int idx = tid + i * 128;
    int r = idx >> 3, ch = idx & 7;
    *(v8h*)&Qs[r][ch * 8] = *(const v8h*)(qb + (size_t)r * DH_ + ch * 8);
  }
  __syncthreads();  // Q visible to all waves

  v16h aq[2];
#pragma unroll
  for (int ks = 0; ks < 2; ++ks) {
    int r = wave * 16 + c;
    aq[ks] = frag2(&Qs[r][32 * ks + 8 * hl], &Qs[r][32 * ks + 16 + 8 * hl]);
  }

  v8f oacc[4];
#pragma unroll
  for (int t = 0; t < 4; ++t) oacc[t] = zero8();
  float mrun[8], lrun[8];
#pragma unroll
  for (int j = 0; j < 8; ++j) { mrun[j] = -3.0e30f; lrun[j] = 0.0f; }
  const float scale = 0.125f;  // 1/sqrt(64)

  int buf = 0;
  for (int kb = 0; kb < S_; kb += 64, buf ^= 1) {
    async_wait0();     // this wave's pending async fills of Ks/Vs[buf] done
    __syncthreads();   // => all waves' fills done; prior reads of buf^1 done

    // Stream next key block into the other buffer while computing this one.
    int kn = kb + 64;
    if (kn < S_) {
      const half_t* kbp = kbase + (size_t)kn * DH_;
      const half_t* vbp = vbase + kn;
#pragma unroll
      for (int i = 0; i < 4; ++i) {
        int idx = tid + i * 128;
        int r = idx >> 3, ch = idx & 7;
        async_copy16(kbp + (size_t)r * DH_ + ch * 8, &Ks[buf ^ 1][r][ch * 8]);
        async_copy16(vbp + (size_t)r * S_ + ch * 8, &Vs[buf ^ 1][r][ch * 8]);
      }
    }

    // scores: 16 x 64, contraction over DH=64 (2 k-steps)
    v8f st[4];
#pragma unroll
    for (int t = 0; t < 4; ++t) st[t] = zero8();
#pragma unroll
    for (int ks = 0; ks < 2; ++ks)
#pragma unroll
      for (int t = 0; t < 4; ++t) {
        v16h bk = frag2(&Ks[buf][t * 16 + c][32 * ks + 8 * hl],
                        &Ks[buf][t * 16 + c][32 * ks + 16 + 8 * hl]);
        st[t] = wmma_f16(aq[ks], bk, st[t]);
      }

    float ma[4];
#pragma unroll
    for (int t = 0; t < 4; ++t) ma[t] = MA[b * S_ + kb + t * 16 + c];

    float p[4][8];
#pragma unroll
    for (int j = 0; j < 8; ++j) {
      float mx = -3.0e30f;
#pragma unroll
      for (int t = 0; t < 4; ++t) {
        float x = st[t][j] * scale + ma[t];
        p[t][j] = x;
        mx = fmaxf(mx, x);
      }
      mx = rmax16(mx);                       // row max across 16 lanes
      float mnew = fmaxf(mrun[j], mx);
      float corr = __expf(mrun[j] - mnew);
      float rs = 0.0f;
#pragma unroll
      for (int t = 0; t < 4; ++t) {
        p[t][j] = __expf(p[t][j] - mnew);
        rs += p[t][j];
      }
      rs = rsum16(rs);
      lrun[j] = lrun[j] * corr + rs;
      mrun[j] = mnew;
#pragma unroll
      for (int t = 0; t < 4; ++t) oacc[t][j] *= corr;
    }

    // C-layout -> A-layout via per-wave LDS buffer (LDS in-order per wave)
#pragma unroll
    for (int t = 0; t < 4; ++t)
#pragma unroll
      for (int j = 0; j < 8; ++j)
        Ps[wave][8 * hl + j][t * 16 + c] = (half_t)p[t][j];

#pragma unroll
    for (int ks = 0; ks < 2; ++ks) {
      v16h ap = frag2(&Ps[wave][c][32 * ks + 8 * hl],
                      &Ps[wave][c][32 * ks + 16 + 8 * hl]);
#pragma unroll
      for (int t = 0; t < 4; ++t) {
        v16h bv = frag2(&Vs[buf][t * 16 + c][32 * ks + 8 * hl],
                        &Vs[buf][t * 16 + c][32 * ks + 16 + 8 * hl]);
        oacc[t] = wmma_f16(ap, bv, oacc[t]);
      }
    }
  }

  // normalize + store ctx (merged heads, f16)
#pragma unroll
  for (int j = 0; j < 8; ++j) {
    float inv = 1.0f / lrun[j];
    int s = s0 + wave * 16 + 8 * hl + j;
#pragma unroll
    for (int t = 0; t < 4; ++t) {
      int dh = t * 16 + c;
      CTX[((size_t)(b * S_ + s)) * D_ + h * DH_ + dh] = (half_t)(oacc[t][j] * inv);
    }
  }
}

// ---------------------------------------------------------------------------
// Output projection: out[m][n] = sum_k ctx16[m][k] * Wo16[n][k], f32 store.
// ---------------------------------------------------------------------------
__global__ __launch_bounds__(256) void oproj_kernel(const half_t* __restrict__ Xh,
                                                    const half_t* __restrict__ W,
                                                    float* __restrict__ out) {
  __shared__ half_t As[128][40];
  __shared__ half_t Bs[128][40];

  const int tid = threadIdx.x;
  const int lane = tid & 31, wave = tid >> 5;
  const int c = lane & 15, hl = lane >> 4;
  const int wm = wave & 3, wn = wave >> 2;
  const int mb = blockIdx.y * 128, nb = blockIdx.x * 128;

  v8f acc[2][4];
#pragma unroll
  for (int i = 0; i < 2; ++i)
#pragma unroll
    for (int t = 0; t < 4; ++t) acc[i][t] = zero8();

  for (int k0 = 0; k0 < D_; k0 += 32) {
    __syncthreads();
#pragma unroll
    for (int i = 0; i < 2; ++i) {
      int idx = tid + i * 256;
      int r = idx >> 2, ch = idx & 3;
      async_copy16(Xh + (size_t)(mb + r) * D_ + k0 + ch * 8, &As[r][ch * 8]);
      async_copy16(W + (size_t)(nb + r) * D_ + k0 + ch * 8, &Bs[r][ch * 8]);
    }
    async_wait0();
    __syncthreads();

    v16h af[2], bf[4];
#pragma unroll
    for (int i = 0; i < 2; ++i) {
      int r = wm * 32 + i * 16 + c;
      af[i] = frag2(&As[r][8 * hl], &As[r][16 + 8 * hl]);
    }
#pragma unroll
    for (int t = 0; t < 4; ++t) {
      int r = wn * 64 + t * 16 + c;
      bf[t] = frag2(&Bs[r][8 * hl], &Bs[r][16 + 8 * hl]);
    }
#pragma unroll
    for (int i = 0; i < 2; ++i)
#pragma unroll
      for (int t = 0; t < 4; ++t) acc[i][t] = wmma_f16(af[i], bf[t], acc[i][t]);
  }

#pragma unroll
  for (int i = 0; i < 2; ++i)
#pragma unroll
    for (int t = 0; t < 4; ++t)
#pragma unroll
      for (int j = 0; j < 8; ++j) {
        int rg = mb + wm * 32 + i * 16 + 8 * hl + j;
        int cg = nb + wn * 64 + t * 16 + c;
        out[(size_t)rg * D_ + cg] = acc[i][t][j];
      }
}

// ---------------------------------------------------------------------------
extern "C" void kernel_launch(void* const* d_in, const int* in_sizes, int n_in,
                              void* d_out, int out_size, void* d_ws, size_t ws_size,
                              hipStream_t stream) {
  const float* q = (const float*)d_in[0];
  const float* k = (const float*)d_in[1];
  const float* v = (const float*)d_in[2];
  const unsigned char* mask = (const unsigned char*)d_in[3];  // jnp.bool_ = 1 byte
  const float* Wq = (const float*)d_in[4];
  const float* Wk = (const float*)d_in[5];
  const float* Wv = (const float*)d_in[6];
  const float* Wo = (const float*)d_in[7];

  char* ws = (char*)d_ws;
  const size_t DDb = (size_t)D_ * D_;            // 1M elems
  half_t* w16  = (half_t*)(ws);                              // 4*DDb f16
  half_t* q16  = (half_t*)(ws + 8ull * 1024 * 1024);
  half_t* k16  = (half_t*)(ws + 24ull * 1024 * 1024);
  half_t* vt16 = (half_t*)(ws + 40ull * 1024 * 1024);
  half_t* c16  = (half_t*)(ws + 56ull * 1024 * 1024);
  float*  mAdd = (float*)(ws + 72ull * 1024 * 1024);         // BS_ f32

  // weight + mask conversion
  const int n4 = (int)(DDb / 4);
  cvt_f32_to_f16<<<(n4 + 255) / 256, 256, 0, stream>>>(Wq, w16 + 0 * DDb, n4);
  cvt_f32_to_f16<<<(n4 + 255) / 256, 256, 0, stream>>>(Wk, w16 + 1 * DDb, n4);
  cvt_f32_to_f16<<<(n4 + 255) / 256, 256, 0, stream>>>(Wv, w16 + 2 * DDb, n4);
  cvt_f32_to_f16<<<(n4 + 255) / 256, 256, 0, stream>>>(Wo, w16 + 3 * DDb, n4);
  cvt_mask<<<(BS_ + 255) / 256, 256, 0, stream>>>(mask, mAdd, BS_);

  // Q/K/V projections (WMMA GEMM)
  dim3 pg(D_ / 128, BS_ / 128);  // (8, 64)
  proj_kernel<<<pg, 256, 0, stream>>>(q, w16 + 0 * DDb, q16, 0);
  proj_kernel<<<pg, 256, 0, stream>>>(k, w16 + 1 * DDb, k16, 1);
  proj_kernel<<<pg, 256, 0, stream>>>(v, w16 + 2 * DDb, vt16, 2);

  // flash attention
  dim3 ag(S_ / 64, B_ * H_);  // (32, 64)
  attn_kernel<<<ag, 128, 0, stream>>>(q16, k16, vt16, mAdd, c16);

  // output projection -> f32 d_out
  oproj_kernel<<<pg, 256, 0, stream>>>(c16, w16 + 3 * DDb, (float*)d_out);

  (void)in_sizes; (void)n_in; (void)out_size; (void)ws_size;
}
